// DecoderLayer_53549652246889
// MI455X (gfx1250) — compile-verified, matrix operand order checked by reading, not run on hardware
//
#include <hip/hip_runtime.h>

typedef _Float16 half_t;
typedef unsigned int u32;
typedef unsigned long long u64;
typedef __attribute__((ext_vector_type(16))) _Float16 v16h;
typedef __attribute__((ext_vector_type(8)))  _Float16 v8h;
typedef __attribute__((ext_vector_type(4)))  _Float16 v4h;
typedef __attribute__((ext_vector_type(8)))  float    v8f;
typedef __attribute__((ext_vector_type(4))) u32 u32x4;
typedef __attribute__((ext_vector_type(8))) int i32x8;
typedef __attribute__((ext_vector_type(4))) int i32x4;

// ---------------------------------------------------------------------------
// WMMA helpers (CDNA5 gfx1250, wave32)
// ---------------------------------------------------------------------------
static __device__ __forceinline__ v8f wmma_f16(v16h a, v16h b, v8f c) {
  return __builtin_amdgcn_wmma_f32_16x16x32_f16(
      /*neg_a=*/false, a, /*neg_b=*/false, b,
      /*c_mod=*/(short)0, c, /*reuse_a=*/false, /*reuse_b=*/false);
}

static __device__ __forceinline__ v8f vzero8f() {
  v8f z;
#pragma unroll
  for (int i = 0; i < 8; ++i) z[i] = 0.0f;
  return z;
}

// 16x32 f16 operand fragment in A layout (== B^T layout) from LDS row-major
// tile with row stride ld.
static __device__ __forceinline__ v16h load_frag(const half_t* p, int ld) {
  const int lane = threadIdx.x & 31;
  const int r    = lane & 15;
  const int sel  = (lane >> 4) * 8;
  const half_t* row = p + r * ld;
  v8h lo = *(const v8h*)(row + sel);
  v8h hi = *(const v8h*)(row + 16 + sel);
  return __builtin_shufflevector(lo, hi,
      0, 1, 2, 3, 4, 5, 6, 7, 8, 9, 10, 11, 12, 13, 14, 15);
}

// ---------------------------------------------------------------------------
// Tensor Data Mover: 2D tile (rows x 32 f16) -> LDS with 8-f16 row padding.
// ---------------------------------------------------------------------------
#if __has_builtin(__builtin_amdgcn_tensor_load_to_lds) && \
    __has_builtin(__builtin_amdgcn_s_wait_tensorcnt)
#define HAVE_TDM 1
#else
#define HAVE_TDM 0
#endif

#if HAVE_TDM
static __device__ __forceinline__ void tdm_load_tile_2d(
    u64 gaddr, u32 lds_off, u32 tensor_d0, u32 tensor_d1,
    u32 tile_d0, u32 tile_d1, u64 d0_stride) {
  u32x4 g0;
  g0[0] = 1u;                                         // count=1, user mode
  g0[1] = lds_off;                                    // lds_addr
  g0[2] = (u32)(gaddr & 0xffffffffu);                 // global_addr[31:0]
  g0[3] = (u32)((gaddr >> 32) & 0x01ffffffu)          // global_addr[56:32]
        | (2u << 30);                                 // type = 2 ("image")
  i32x8 g1;
  // data_size=1 (2B) | pad_enable | pad_interval=3 (16 DW) | pad_amount=3 (4 DW)
  g1[0] = (int)((1u << 16) | (1u << 20) | (3u << 22) | (3u << 25));
  g1[1] = (int)((tensor_d0 & 0xffffu) << 16);
  g1[2] = (int)(((tensor_d0 >> 16) & 0xffffu) | ((tensor_d1 & 0xffffu) << 16));
  g1[3] = (int)(((tensor_d1 >> 16) & 0xffffu) | ((tile_d0 & 0xffffu) << 16));
  g1[4] = (int)(tile_d1 & 0xffffu);
  g1[5] = (int)(u32)(d0_stride & 0xffffffffu);
  g1[6] = (int)(u32)((d0_stride >> 32) & 0xffffu);
  g1[7] = 0;
  i32x4 z4;
  z4[0] = 0; z4[1] = 0; z4[2] = 0; z4[3] = 0;
#if defined(__clang_major__) && (__clang_major__ >= 23)
  i32x8 z8;
#pragma unroll
  for (int i = 0; i < 8; ++i) z8[i] = 0;
  __builtin_amdgcn_tensor_load_to_lds(g0, g1, z4, z4, z8, 0);
#else
  __builtin_amdgcn_tensor_load_to_lds(g0, g1, z4, z4, 0);
#endif
}

static __device__ __forceinline__ u32 lds_offset_of(const void* p) {
  return (u32)(uintptr_t)(const __attribute__((address_space(3))) void*)p;
}
#endif  // HAVE_TDM

// ---------------------------------------------------------------------------
// fp32 -> f16 conversion
// ---------------------------------------------------------------------------
__global__ __launch_bounds__(256)
void convert_to_f16_kernel(const float* __restrict__ in, half_t* __restrict__ out,
                           int n) {
  int i = (blockIdx.x * 256 + threadIdx.x) * 4;
  if (i < n) {
    float4 v = *(const float4*)(in + i);
    v4h o;
    o[0] = (half_t)v.x; o[1] = (half_t)v.y;
    o[2] = (half_t)v.z; o[3] = (half_t)v.w;
    *(v4h*)(out + i) = o;
  }
}

// fp32 [K][N] -> f16 [N][K]
__global__ void transpose_to_f16_kernel(const float* __restrict__ in,
                                        half_t* __restrict__ out, int K, int N) {
  __shared__ float tile[32][33];
  const int n0 = blockIdx.x * 32;
  const int k0 = blockIdx.y * 32;
  const int tx = threadIdx.x;
  const int ty = threadIdx.y;
#pragma unroll
  for (int i = 0; i < 4; ++i)
    tile[ty + i * 8][tx] = in[(size_t)(k0 + ty + i * 8) * N + n0 + tx];
  __syncthreads();
#pragma unroll
  for (int i = 0; i < 4; ++i)
    out[(size_t)(n0 + ty + i * 8) * K + k0 + tx] = (half_t)tile[tx][ty + i * 8];
}

// ---------------------------------------------------------------------------
// WMMA GEMM: out = act( A[M,K] * Bt[N,K]^T + bias (+ residual) )
// Workgroup tile 256x128, 8 waves (4x2), 64x64 per wave; double-buffered LDS;
// operand streaming via Tensor Data Mover when available.
// ---------------------------------------------------------------------------
template <bool RES, bool RELU, bool O32, bool O16>
__global__ __launch_bounds__(256)
void gemm_f16_kernel(const half_t* __restrict__ A, const half_t* __restrict__ Bt,
                     const float* __restrict__ bias, const float* __restrict__ resid,
                     float* __restrict__ out32, half_t* __restrict__ out16,
                     int M, int N, int K) {
  __shared__ __align__(16) half_t As[2][256][40];
  __shared__ __align__(16) half_t Bs[2][128][40];
  const int tid = threadIdx.x;
  const int m0 = blockIdx.y * 256;
  const int n0 = blockIdx.x * 128;
  const int wave = tid >> 5;
  const int wm = (wave >> 1) * 64;
  const int wn = (wave & 1) * 64;

  v8f acc[4][4];
#pragma unroll
  for (int i = 0; i < 4; ++i)
#pragma unroll
    for (int j = 0; j < 4; ++j) acc[i][j] = vzero8f();

#if HAVE_TDM
  if (tid < 32) {
    tdm_load_tile_2d((u64)(uintptr_t)(A + (size_t)m0 * K),
                     lds_offset_of(&As[0][0][0]), (u32)K, (u32)M, 32u, 256u,
                     (u64)(u32)K);
    tdm_load_tile_2d((u64)(uintptr_t)(Bt + (size_t)n0 * K),
                     lds_offset_of(&Bs[0][0][0]), (u32)K, (u32)N, 32u, 128u,
                     (u64)(u32)K);
    __builtin_amdgcn_s_wait_tensorcnt(0);
  }
  __syncthreads();

  int buf = 0;
  for (int kk = 0; kk < K; kk += 32) {
    const bool has_next = (kk + 32) < K;
    if (has_next && tid < 32) {
      tdm_load_tile_2d((u64)(uintptr_t)(A + (size_t)m0 * K + kk + 32),
                       lds_offset_of(&As[buf ^ 1][0][0]), (u32)K, (u32)M, 32u,
                       256u, (u64)(u32)K);
      tdm_load_tile_2d((u64)(uintptr_t)(Bt + (size_t)n0 * K + kk + 32),
                       lds_offset_of(&Bs[buf ^ 1][0][0]), (u32)K, (u32)N, 32u,
                       128u, (u64)(u32)K);
    }
    v16h af[4];
#pragma unroll
    for (int mt = 0; mt < 4; ++mt)
      af[mt] = load_frag(&As[buf][wm + mt * 16][0], 40);
#pragma unroll
    for (int nt = 0; nt < 4; ++nt) {
      v16h bf = load_frag(&Bs[buf][wn + nt * 16][0], 40);
#pragma unroll
      for (int mt = 0; mt < 4; ++mt)
        acc[mt][nt] = wmma_f16(af[mt], bf, acc[mt][nt]);
    }
    if (has_next) {
      if (tid < 32) __builtin_amdgcn_s_wait_tensorcnt(0);
      __syncthreads();
      buf ^= 1;
    }
  }
#else
  uint4 pa[4], pb[2];
  auto gload = [&](int kk) {
#pragma unroll
    for (int i = 0; i < 4; ++i) {
      int idx = tid + i * 256;
      pa[i] = *(const uint4*)(A + (size_t)(m0 + (idx >> 2)) * K + kk +
                              (idx & 3) * 8);
    }
#pragma unroll
    for (int i = 0; i < 2; ++i) {
      int idx = tid + i * 256;
      pb[i] = *(const uint4*)(Bt + (size_t)(n0 + (idx >> 2)) * K + kk +
                              (idx & 3) * 8);
    }
  };
  auto lstore = [&](int b) {
#pragma unroll
    for (int i = 0; i < 4; ++i) {
      int idx = tid + i * 256;
      *(uint4*)&As[b][idx >> 2][(idx & 3) * 8] = pa[i];
    }
#pragma unroll
    for (int i = 0; i < 2; ++i) {
      int idx = tid + i * 256;
      *(uint4*)&Bs[b][idx >> 2][(idx & 3) * 8] = pb[i];
    }
  };
  gload(0);
  lstore(0);
  __syncthreads();
  int buf = 0;
  for (int kk = 0; kk < K; kk += 32) {
    const bool has_next = (kk + 32) < K;
    if (has_next) gload(kk + 32);
    v16h af[4];
#pragma unroll
    for (int mt = 0; mt < 4; ++mt)
      af[mt] = load_frag(&As[buf][wm + mt * 16][0], 40);
#pragma unroll
    for (int nt = 0; nt < 4; ++nt) {
      v16h bf = load_frag(&Bs[buf][wn + nt * 16][0], 40);
#pragma unroll
      for (int mt = 0; mt < 4; ++mt)
        acc[mt][nt] = wmma_f16(af[mt], bf, acc[mt][nt]);
    }
    if (has_next) {
      lstore(buf ^ 1);
      __syncthreads();
      buf ^= 1;
    }
  }
#endif

  const int lane = tid & 31;
  const int nlo = lane & 15;
  const int mhi = (lane >> 4) * 8;
#pragma unroll
  for (int mt = 0; mt < 4; ++mt) {
#pragma unroll
    for (int nt = 0; nt < 4; ++nt) {
      const int gcol = n0 + wn + nt * 16 + nlo;
      const float bv = bias[gcol];
#pragma unroll
      for (int j = 0; j < 8; ++j) {
        const int grow = m0 + wm + mt * 16 + mhi + j;
        float v = acc[mt][nt][j] + bv;
        if (RES) v += resid[(size_t)grow * N + gcol];
        if (RELU) v = fmaxf(v, 0.0f);
        if (O32) out32[(size_t)grow * N + gcol] = v;
        if (O16) out16[(size_t)grow * N + gcol] = (half_t)v;
      }
    }
  }
}

// ---------------------------------------------------------------------------
// Flash attention (online softmax), head_dim=64, 64 q-rows per workgroup,
// 4 waves x 16 rows. Q pre-scaled by 1/sqrt(64) (exact in f16). Double-
// buffered K/V tiles: one workgroup barrier per k-block. P restaged through
// wave-private LDS (wavefront-scope fence only).
// ---------------------------------------------------------------------------
template <bool CAUSAL>
__global__ __launch_bounds__(128)
void flash_attn_kernel(const half_t* __restrict__ Q, const half_t* __restrict__ Kg,
                       const half_t* __restrict__ Vg, half_t* __restrict__ Out,
                       int S, int nheads) {
  __shared__ __align__(16) half_t Qs[64][72];
  __shared__ __align__(16) half_t Ks[2][64][72];    // [key][d]
  __shared__ __align__(16) half_t Vt[2][64][72];    // [d][key]
  __shared__ __align__(16) half_t Ps[4][16][72];    // wave-private P staging

  const int dm   = nheads * 64;
  const int tid  = threadIdx.x;
  const int qt   = blockIdx.x;
  const int b    = blockIdx.y / nheads;
  const int h    = blockIdx.y % nheads;
  const size_t tokbase = (size_t)b * S;
  const int fbase = h * 64;
  const int w    = tid >> 5;
  const int lane = tid & 31;
  const int nlo  = lane & 15;
  const int mhi  = (lane >> 4) * 8;

  // load Q tile, pre-scaled by 0.125 (exact power-of-two scale)
#pragma unroll
  for (int i = 0; i < 4; ++i) {
    int idx = tid + i * 128;
    int r = idx >> 3;
    int c = (idx & 7) * 8;
    v8h q = *(const v8h*)(Q + (tokbase + qt * 64 + r) * dm + fbase + c);
#pragma unroll
    for (int j = 0; j < 8; ++j) q[j] = q[j] * (half_t)0.125f;
    *(v8h*)&Qs[r][c] = q;
  }

  uint4 kreg[4];
  v8h vreg[4];
  auto gload_kv = [&](int kt) {
#pragma unroll
    for (int i = 0; i < 4; ++i) {
      int idx = tid + i * 128;
      int r = idx >> 3;
      int c = (idx & 7) * 8;
      const size_t base = (tokbase + kt * 64 + r) * dm + fbase + c;
      kreg[i] = *(const uint4*)(Kg + base);
      vreg[i] = *(const v8h*)(Vg + base);
    }
  };
  auto lstore_kv = [&](int bb) {
#pragma unroll
    for (int i = 0; i < 4; ++i) {
      int idx = tid + i * 128;
      int r = idx >> 3;
      int c = (idx & 7) * 8;
      *(uint4*)&Ks[bb][r][c] = kreg[i];
#pragma unroll
      for (int j = 0; j < 8; ++j) Vt[bb][c + j][r] = vreg[i][j];
    }
  };

  gload_kv(0);
  lstore_kv(0);
  __syncthreads();

  const v16h aq0 = load_frag(&Qs[w * 16][0], 72);
  const v16h aq1 = load_frag(&Qs[w * 16][32], 72);

  v8f o[4];
#pragma unroll
  for (int dt = 0; dt < 4; ++dt) o[dt] = vzero8f();
  float mx[8], sm[8];
#pragma unroll
  for (int j = 0; j < 8; ++j) { mx[j] = -1e30f; sm[j] = 0.0f; }

  const int kend = CAUSAL ? (qt + 1) : (S >> 6);
  int buf = 0;
  for (int kt = 0; kt < kend; ++kt) {
    const bool has_next = (kt + 1) < kend;
    if (has_next) gload_kv(kt + 1);

    v8f s[4];
#pragma unroll
    for (int nt = 0; nt < 4; ++nt) {
      v8f t = vzero8f();
      t = wmma_f16(aq0, load_frag(&Ks[buf][nt * 16][0], 72), t);
      t = wmma_f16(aq1, load_frag(&Ks[buf][nt * 16][32], 72), t);
      s[nt] = t;
    }
    if (CAUSAL && kt == qt) {   // diagonal block only (uniform branch)
#pragma unroll
      for (int nt = 0; nt < 4; ++nt)
#pragma unroll
        for (int j = 0; j < 8; ++j)
          if (nt * 16 + nlo > w * 16 + mhi + j) s[nt][j] = -1e30f;
    }
    float alpha[8];
#pragma unroll
    for (int j = 0; j < 8; ++j) {
      float v = fmaxf(fmaxf(s[0][j], s[1][j]), fmaxf(s[2][j], s[3][j]));
#pragma unroll
      for (int off = 1; off < 16; off <<= 1) v = fmaxf(v, __shfl_xor(v, off, 32));
      float nm = fmaxf(mx[j], v);
      alpha[j] = __expf(mx[j] - nm);
      mx[j] = nm;
      sm[j] *= alpha[j];
    }
#pragma unroll
    for (int nt = 0; nt < 4; ++nt)
#pragma unroll
      for (int j = 0; j < 8; ++j) s[nt][j] = __expf(s[nt][j] - mx[j]);
#pragma unroll
    for (int j = 0; j < 8; ++j) {
      float v = s[0][j] + s[1][j] + s[2][j] + s[3][j];
#pragma unroll
      for (int off = 1; off < 16; off <<= 1) v += __shfl_xor(v, off, 32);
      sm[j] += v;
    }
#pragma unroll
    for (int dt = 0; dt < 4; ++dt)
#pragma unroll
      for (int j = 0; j < 8; ++j) o[dt][j] *= alpha[j];

#pragma unroll
    for (int nt = 0; nt < 4; ++nt)
#pragma unroll
      for (int j = 0; j < 8; ++j)
        Ps[w][mhi + j][nt * 16 + nlo] = (half_t)s[nt][j];
    // wave-private region: only need wave-scope LDS ordering, not a barrier
    __builtin_amdgcn_fence(__ATOMIC_ACQ_REL, "wavefront");

    const v16h ap0 = load_frag(&Ps[w][0][0], 72);
    const v16h ap1 = load_frag(&Ps[w][0][32], 72);
#pragma unroll
    for (int dt = 0; dt < 4; ++dt) {
      o[dt] = wmma_f16(ap0, load_frag(&Vt[buf][dt * 16][0], 72), o[dt]);
      o[dt] = wmma_f16(ap1, load_frag(&Vt[buf][dt * 16][32], 72), o[dt]);
    }

    if (has_next) {
      lstore_kv(buf ^ 1);
      __syncthreads();
      buf ^= 1;
    }
  }

#pragma unroll
  for (int dt = 0; dt < 4; ++dt)
#pragma unroll
    for (int j = 0; j < 8; ++j) {
      float v = o[dt][j] / sm[j];
      Out[(tokbase + qt * 64 + w * 16 + mhi + j) * dm + fbase + dt * 16 + nlo] =
          (half_t)v;
    }
}

// ---------------------------------------------------------------------------
// Row LayerNorm (N=1024), fp32 out (+ optional f16 copy)
// ---------------------------------------------------------------------------
__global__ __launch_bounds__(256)
void layernorm_kernel(const float* __restrict__ in, const float* __restrict__ gamma,
                      const float* __restrict__ beta, float* __restrict__ out32,
                      half_t* __restrict__ out16, int N) {
  __shared__ float sbuf[8], ssbuf[8];
  const int row = blockIdx.x;
  const float* p = in + (size_t)row * N;
  float s = 0.0f, ss = 0.0f;
  for (int i = threadIdx.x; i < N; i += 256) {
    float v = p[i];
    s += v;
    ss += v * v;
  }
#pragma unroll
  for (int off = 16; off > 0; off >>= 1) {
    s  += __shfl_xor(s, off, 32);
    ss += __shfl_xor(ss, off, 32);
  }
  if ((threadIdx.x & 31) == 0) {
    sbuf[threadIdx.x >> 5]  = s;
    ssbuf[threadIdx.x >> 5] = ss;
  }
  __syncthreads();
  float ts = 0.0f, tss = 0.0f;
#pragma unroll
  for (int wv = 0; wv < 8; ++wv) { ts += sbuf[wv]; tss += ssbuf[wv]; }
  const float mu  = ts / N;
  const float var = tss / N - mu * mu;
  const float r   = rsqrtf(var + 1e-6f);
  for (int i = threadIdx.x; i < N; i += 256) {
    float v = (p[i] - mu) * r * gamma[i] + beta[i];
    out32[(size_t)row * N + i] = v;
    if (out16) out16[(size_t)row * N + i] = (half_t)v;
  }
}

// ---------------------------------------------------------------------------
// Host-side orchestration
// ---------------------------------------------------------------------------
extern "C" void kernel_launch(void* const* d_in, const int* in_sizes, int n_in,
                              void* d_out, int out_size, void* d_ws, size_t ws_size,
                              hipStream_t stream) {
  (void)in_sizes; (void)n_in; (void)out_size; (void)ws_size;
  const int Bn = 4, S = 2048, D = 1024, FF = 4096, H = 16;
  const int M = Bn * S;

  const float* x    = (const float*)d_in[0];
  const float* enc  = (const float*)d_in[1];
  const float* wq1  = (const float*)d_in[2];  const float* bq1 = (const float*)d_in[3];
  const float* wk1  = (const float*)d_in[4];  const float* bk1 = (const float*)d_in[5];
  const float* wv1  = (const float*)d_in[6];  const float* bv1 = (const float*)d_in[7];
  const float* wo1  = (const float*)d_in[8];  const float* bo1 = (const float*)d_in[9];
  const float* wq2  = (const float*)d_in[10]; const float* bq2 = (const float*)d_in[11];
  const float* wk2  = (const float*)d_in[12]; const float* bk2 = (const float*)d_in[13];
  const float* wv2  = (const float*)d_in[14]; const float* bv2 = (const float*)d_in[15];
  const float* wo2  = (const float*)d_in[16]; const float* bo2 = (const float*)d_in[17];
  const float* wf1  = (const float*)d_in[18]; const float* bf1 = (const float*)d_in[19];
  const float* wf2  = (const float*)d_in[20]; const float* bf2 = (const float*)d_in[21];
  const float* g1   = (const float*)d_in[22]; const float* be1 = (const float*)d_in[23];
  const float* g2   = (const float*)d_in[24]; const float* be2 = (const float*)d_in[25];
  const float* g3   = (const float*)d_in[26]; const float* be3 = (const float*)d_in[27];

  char* ws = (char*)d_ws;
  auto alloc = [&](size_t bytes) -> char* {
    char* p = ws;
    ws += (bytes + 255) & ~(size_t)255;
    return p;
  };
  const size_t DDh  = (size_t)D * D * sizeof(half_t);
  const size_t DFFh = (size_t)D * FF * sizeof(half_t);
  const size_t MDh  = (size_t)M * D * sizeof(half_t);
  const size_t MDf  = (size_t)M * D * sizeof(float);

  half_t* wq1t = (half_t*)alloc(DDh);
  half_t* wk1t = (half_t*)alloc(DDh);
  half_t* wv1t = (half_t*)alloc(DDh);
  half_t* wo1t = (half_t*)alloc(DDh);
  half_t* wq2t = (half_t*)alloc(DDh);
  half_t* wk2t = (half_t*)alloc(DDh);
  half_t* wv2t = (half_t*)alloc(DDh);
  half_t* wo2t = (half_t*)alloc(DDh);
  half_t* wf1t = (half_t*)alloc(DFFh);
  half_t* wf2t = (half_t*)alloc(DFFh);
  half_t* xh   = (half_t*)alloc(MDh);
  half_t* ench = (half_t*)alloc(MDh);
  half_t* qb   = (half_t*)alloc(MDh);   // qb..attnb (64MB) doubles as FFN hidden
  half_t* kb   = (half_t*)alloc(MDh);
  half_t* vb   = (half_t*)alloc(MDh);
  half_t* attnb= (half_t*)alloc(MDh);
  float*  tmp32= (float*)alloc(MDf);
  float*  y1   = (float*)alloc(MDf);
  half_t* y1h  = (half_t*)alloc(MDh);
  float*  y2   = (float*)alloc(MDf);
  half_t* y2h  = (half_t*)alloc(MDh);
  half_t* hff  = qb;

  const dim3 tb(32, 8);
  transpose_to_f16_kernel<<<dim3(D / 32, D / 32), tb, 0, stream>>>(wq1, wq1t, D, D);
  transpose_to_f16_kernel<<<dim3(D / 32, D / 32), tb, 0, stream>>>(wk1, wk1t, D, D);
  transpose_to_f16_kernel<<<dim3(D / 32, D / 32), tb, 0, stream>>>(wv1, wv1t, D, D);
  transpose_to_f16_kernel<<<dim3(D / 32, D / 32), tb, 0, stream>>>(wo1, wo1t, D, D);
  transpose_to_f16_kernel<<<dim3(D / 32, D / 32), tb, 0, stream>>>(wq2, wq2t, D, D);
  transpose_to_f16_kernel<<<dim3(D / 32, D / 32), tb, 0, stream>>>(wk2, wk2t, D, D);
  transpose_to_f16_kernel<<<dim3(D / 32, D / 32), tb, 0, stream>>>(wv2, wv2t, D, D);
  transpose_to_f16_kernel<<<dim3(D / 32, D / 32), tb, 0, stream>>>(wo2, wo2t, D, D);
  transpose_to_f16_kernel<<<dim3(FF / 32, D / 32), tb, 0, stream>>>(wf1, wf1t, D, FF);
  transpose_to_f16_kernel<<<dim3(D / 32, FF / 32), tb, 0, stream>>>(wf2, wf2t, FF, D);

  convert_to_f16_kernel<<<(M * D) / 1024, 256, 0, stream>>>(x, xh, M * D);
  convert_to_f16_kernel<<<(M * D) / 1024, 256, 0, stream>>>(enc, ench, M * D);

  const dim3 gD(D / 128, M / 256);
  const dim3 gF(FF / 128, M / 256);
  const dim3 gA(S / 64, Bn * H);

  // ---- masked self-attention ----
  gemm_f16_kernel<false, false, false, true><<<gD, 256, 0, stream>>>(
      xh, wq1t, bq1, nullptr, nullptr, qb, M, D, D);
  gemm_f16_kernel<false, false, false, true><<<gD, 256, 0, stream>>>(
      xh, wk1t, bk1, nullptr, nullptr, kb, M, D, D);
  gemm_f16_kernel<false, false, false, true><<<gD, 256, 0, stream>>>(
      xh, wv1t, bv1, nullptr, nullptr, vb, M, D, D);
  flash_attn_kernel<true><<<gA, 128, 0, stream>>>(qb, kb, vb, attnb, S, H);
  gemm_f16_kernel<true, false, true, false><<<gD, 256, 0, stream>>>(
      attnb, wo1t, bo1, x, tmp32, nullptr, M, D, D);
  layernorm_kernel<<<M, 256, 0, stream>>>(tmp32, g1, be1, y1, y1h, D);

  // ---- cross attention (q=enc, k=enc, v=y1 per reference) ----
  gemm_f16_kernel<false, false, false, true><<<gD, 256, 0, stream>>>(
      ench, wq2t, bq2, nullptr, nullptr, qb, M, D, D);
  gemm_f16_kernel<false, false, false, true><<<gD, 256, 0, stream>>>(
      ench, wk2t, bk2, nullptr, nullptr, kb, M, D, D);
  gemm_f16_kernel<false, false, false, true><<<gD, 256, 0, stream>>>(
      y1h, wv2t, bv2, nullptr, nullptr, vb, M, D, D);
  flash_attn_kernel<false><<<gA, 128, 0, stream>>>(qb, kb, vb, attnb, S, H);
  gemm_f16_kernel<true, false, true, false><<<gD, 256, 0, stream>>>(
      attnb, wo2t, bo2, y1, tmp32, nullptr, M, D, D);
  layernorm_kernel<<<M, 256, 0, stream>>>(tmp32, g2, be2, y2, y2h, D);

  // ---- feed-forward ----
  gemm_f16_kernel<false, true, false, true><<<gF, 256, 0, stream>>>(
      y2h, wf1t, bf1, nullptr, nullptr, hff, M, FF, D);
  gemm_f16_kernel<true, false, true, false><<<gD, 256, 0, stream>>>(
      hff, wf2t, bf2, y2, tmp32, nullptr, M, D, FF);
  layernorm_kernel<<<M, 256, 0, stream>>>(tmp32, g3, be3, (float*)d_out, nullptr, D);
}
